// RNN_54606214202223
// MI455X (gfx1250) — matrix operand order repaired and checked by reading, not compiled, not run on hardware
//
#include <hip/hip_runtime.h>
#include <hip/hip_bf16.h>

// ---------------------------------------------------------------------------
// 2-layer tanh RNN on gfx1250 (MI455X): bf16 WMMA + async-LDS blocked GEMM.
// Block = 256 threads (8 wave32) computes a 64(M) x 64(N) output tile.
// K is staged in 128-wide double-buffered LDS panels via the CDNA5 async
// global->LDS engine (ASYNCcnt); WMMA fragments are read with ds_load_b128.
// Per-thread staging pointers are precomputed so the per-stage address math
// is add-only (no multiplies in the pipelined loop).
// ---------------------------------------------------------------------------

typedef __bf16 bf16_t;
typedef __attribute__((ext_vector_type(16))) __bf16 v16bf;
typedef __attribute__((ext_vector_type(8)))  __bf16 v8bf;
typedef __attribute__((ext_vector_type(8)))  float  v8f;
typedef __attribute__((ext_vector_type(4)))  int    v4i;

#define RNN_HID   2048
#define RNN_EMB   2048
#define RNN_VOCAB 10000
#define RNN_SEQ   128
#define RNN_BATCH 64
#define KC        128   // staged K panel width

// ------------------- CDNA5 async global->LDS availability -------------------
#if __has_builtin(__builtin_amdgcn_global_load_async_to_lds_b128) && \
    __has_builtin(__builtin_amdgcn_s_wait_asynccnt)
#define RNN_ASYNC 1
#else
#define RNN_ASYNC 0
#endif

#if RNN_ASYNC
typedef __attribute__((address_space(1))) v4i glb_v4i;
typedef __attribute__((address_space(3))) v4i lds_v4i;
__device__ __forceinline__ void async_copy_b128(const void* g, void* l) {
  // global_load_async_to_lds_b128 (per-lane 16B), tracked by ASYNCcnt
  __builtin_amdgcn_global_load_async_to_lds_b128(
      (glb_v4i*)g, (lds_v4i*)l, /*offset=*/0, /*cpol=*/0);
}
#endif

// ---------------------------------------------------------------------------
// Blocked GEMM:  out[m][n] = act( concat(inA0,inA1)[m][:] . W[n][:] + bias[n] )
//   A: M=64 rows; cols [0,KS) from inA0, [KS,K) from inA1 (row strides KS, K-KS)
//   W: row-major [>=N][K] bf16.  One block -> 64x64 tile of out.
//   8 waves: waveM = wid&3 (M tile), waveNg = wid>>2 (32-col group, 2 WMMA accs)
//   KS, K must be multiples of KC.
// ---------------------------------------------------------------------------
template <bool TANH, bool NGUARD>
__global__ __launch_bounds__(256) void k_gemm_bf16(
    const bf16_t* __restrict__ inA0, const bf16_t* __restrict__ inA1,
    int KS, int K, int N,
    const bf16_t* __restrict__ W, const float* __restrict__ bias,
    float* __restrict__ outF, bf16_t* __restrict__ outBf, int ldOut) {
  __shared__ bf16_t sA[2][64 * KC];   // 16KB per buffer
  __shared__ bf16_t sB[2][64 * KC];   // 16KB per buffer

  const int tid  = threadIdx.x;
  const int lane = tid & 31;
  const int wid  = tid >> 5;
  const int half = lane >> 4;
  const int l16  = lane & 15;
  const int waveM  = wid & 3;         // M tile 0..3
  const int waveNg = wid >> 2;        // N group 0..1 (32 cols each)
  const int nBlock = blockIdx.x * 64; // block's N base
  const int nTile0 = waveNg * 32;     // within-block col base

  // ---- per-thread staging geometry (fixed): 8 x 16B segments per stage ----
  // segment s (A and B each, s=0..3): row = (tid>>4) + 16*s, col = (tid&15)*8
  const int rBase = tid >> 4;         // 0..15
  const int cOff  = (tid & 15) * 8;   // 0..120
  const bf16_t* aBase0[4];            // inA0 row bases (add k0)
  const bf16_t* aBase1[4];            // inA1 row bases (add k0 directly, k0>=KS)
  const bf16_t* bBase[4];             // W row bases (add k0)
#pragma unroll
  for (int s = 0; s < 4; ++s) {
    const int r = rBase + 16 * s;     // 0..63
    aBase0[s] = inA0 + (size_t)r * KS + cOff;
    aBase1[s] = inA1 + ((size_t)r * (size_t)(K - KS) + cOff) - (size_t)KS;
    bBase[s]  = W + (size_t)(nBlock + r) * K + cOff;
  }

  const int nStages = K / KC;

  auto issue = [&](int k0, int buf) {
    const bool ph0 = k0 < KS;         // uniform: which A source this panel uses
    bf16_t* la = &sA[buf][tid * 8];
    bf16_t* lb = &sB[buf][tid * 8];
#pragma unroll
    for (int s = 0; s < 4; ++s) {
      const bf16_t* ga = (ph0 ? aBase0[s] : aBase1[s]) + k0;
      const bf16_t* gb = bBase[s] + k0;
#if RNN_ASYNC
      async_copy_b128(ga, la + s * (16 * KC));
      async_copy_b128(gb, lb + s * (16 * KC));
#else
      *(v8bf*)(la + s * (16 * KC)) = *(const v8bf*)ga;
      *(v8bf*)(lb + s * (16 * KC)) = *(const v8bf*)gb;
#endif
    }
  };

  v8f acc0 = {};
  v8f acc1 = {};

  issue(0, 0);
  for (int s = 0; s < nStages; ++s) {
    const int buf = s & 1;
    if (s + 1 < nStages) issue((s + 1) * KC, buf ^ 1);
#if RNN_ASYNC
    // 8 async ops/thread per stage; stage s done when <= next stage's 8 remain
    if (s + 1 < nStages) __builtin_amdgcn_s_wait_asynccnt(8);
    else                 __builtin_amdgcn_s_wait_asynccnt(0);
#endif
    __syncthreads();

    const bf16_t* la = sA[buf];
    const bf16_t* lb = sB[buf];
#pragma unroll
    for (int c = 0; c < KC / 32; ++c) {   // four 16x16x32 K-chunks per stage
      const int kk = c * 32;
      // A fragment: lane holds row m, 16 contiguous K values (K = 16*half + s)
      const bf16_t* ap = la + (size_t)(waveM * 16 + l16) * KC + kk + half * 16;
      v8bf alo = *(const v8bf*)ap;        // ds_load_b128
      v8bf ahi = *(const v8bf*)(ap + 8);
      v16bf a  = __builtin_shufflevector(alo, ahi,
                   0,1,2,3,4,5,6,7,8,9,10,11,12,13,14,15);
      // B fragments: lane holds column n, 16 contiguous K values
      const bf16_t* bp0 = lb + (size_t)(nTile0 + l16)      * KC + kk + half * 16;
      const bf16_t* bp1 = lb + (size_t)(nTile0 + 16 + l16) * KC + kk + half * 16;
      v16bf b0 = *(const v16bf*)bp0;
      v16bf b1 = *(const v16bf*)bp1;
      acc0 = __builtin_amdgcn_wmma_f32_16x16x32_bf16(false, a, false, b0,
                                                     (short)0, acc0, false, false);
      acc1 = __builtin_amdgcn_wmma_f32_16x16x32_bf16(false, a, false, b1,
                                                     (short)0, acc1, false, false);
    }
    __syncthreads();  // all waves done with buf before it is refilled
  }

  const int n0g = nBlock + nTile0 + l16;   // lane's global column (tile 0)
  const int n1g = n0g + 16;                // lane's global column (tile 1)
  const bool ok0 = !NGUARD || (n0g < N);
  const bool ok1 = !NGUARD || (n1g < N);
  const float bn0 = ok0 ? bias[n0g] : 0.0f;
  const float bn1 = ok1 ? bias[n1g] : 0.0f;

#pragma unroll
  for (int r = 0; r < 8; ++r) {
    const int mo = waveM * 16 + r + 8 * half;  // C/D layout: M = r + 8*(lane/16)
    float x0 = acc0[r] + bn0;
    float x1 = acc1[r] + bn1;
    if (TANH) { x0 = tanhf(x0); x1 = tanhf(x1); }
    const size_t o0 = (size_t)mo * ldOut + n0g;
    const size_t o1 = (size_t)mo * ldOut + n1g;
    if (ok0) {
      if (outF)  outF[o0]  = x0;
      if (outBf) outBf[o0] = (bf16_t)x0;
    }
    if (ok1) {
      if (outF)  outF[o1]  = x1;
      if (outBf) outBf[o1] = (bf16_t)x1;
    }
  }
}

// ----------------------------- helpers -------------------------------------

__global__ __launch_bounds__(256) void k_cvt_f32_bf16(const float* __restrict__ src,
                                                      bf16_t* __restrict__ dst, int n) {
  int i = blockIdx.x * 256 + threadIdx.x;
  if (i < n) dst[i] = (bf16_t)src[i];
}

// Embedding gather for one timestep: x[b][e] = bf16(emb[tok[b]][e])
__global__ __launch_bounds__(256) void k_embed(const int* __restrict__ tok,
                                               const float* __restrict__ emb,
                                               bf16_t* __restrict__ x) {
  int i = blockIdx.x * 256 + threadIdx.x;  // BATCH*EMB threads
  int b = i >> 11;                         // /2048
  int e = i & 2047;
  x[i] = (bf16_t)emb[(long long)tok[b] * RNN_EMB + e];
}

// ------------------------------- driver ------------------------------------

extern "C" void kernel_launch(void* const* d_in, const int* in_sizes, int n_in,
                              void* d_out, int out_size, void* d_ws, size_t ws_size,
                              hipStream_t stream) {
  const int*   inputs = (const int*)  d_in[0];  // [SEQ][BATCH]
  const float* hidden = (const float*)d_in[1];  // [2][BATCH][HID]
  const float* emb    = (const float*)d_in[2];  // [VOCAB][EMB]
  const float* W0     = (const float*)d_in[3];  // [HID][EMB+HID]
  const float* b0     = (const float*)d_in[4];
  const float* W1     = (const float*)d_in[5];  // [HID][2*HID]
  const float* b1     = (const float*)d_in[6];
  const float* Wout   = (const float*)d_in[7];  // [VOCAB][HID]
  const float* bout   = (const float*)d_in[8];
  float* out = (float*)d_out;  // logits [SEQ][BATCH][VOCAB] ++ final_hidden f32

  const size_t nW   = (size_t)RNN_HID * 2 * RNN_HID;  // 8,388,608
  const size_t nWo  = (size_t)RNN_VOCAB * RNN_HID;    // 20,480,000
  const size_t nH   = (size_t)RNN_BATCH * RNN_HID;    // 131,072
  const size_t nLog = (size_t)RNN_BATCH * RNN_VOCAB;  // 640,000 / step

  char* ws = (char*)d_ws;
  size_t off = 0;
  auto carve = [&](size_t bytes) {
    char* p = ws + off; off = (off + bytes + 255) & ~(size_t)255; return p;
  };
  bf16_t* w0bf    = (bf16_t*)carve(nW * 2);
  bf16_t* w1bf    = (bf16_t*)carve(nW * 2);
  bf16_t* wobf    = (bf16_t*)carve(nWo * 2);
  bf16_t* h0bf[2] = {(bf16_t*)carve(nH * 2), (bf16_t*)carve(nH * 2)};
  bf16_t* h1bf[2] = {(bf16_t*)carve(nH * 2), (bf16_t*)carve(nH * 2)};
  bf16_t* xbf     = (bf16_t*)carve(nH * 2);   // one timestep of embeddings
  float*  hf32    = (float*) carve(2 * nH * 4);

  // One-time (per launch) f32 -> bf16 conversions; weights then live in L2.
  k_cvt_f32_bf16<<<(int)((nW  + 255) / 256), 256, 0, stream>>>(W0,   w0bf, (int)nW);
  k_cvt_f32_bf16<<<(int)((nW  + 255) / 256), 256, 0, stream>>>(W1,   w1bf, (int)nW);
  k_cvt_f32_bf16<<<(int)((nWo + 255) / 256), 256, 0, stream>>>(Wout, wobf, (int)nWo);
  k_cvt_f32_bf16<<<(int)((nH  + 255) / 256), 256, 0, stream>>>(hidden,      h0bf[0], (int)nH);
  k_cvt_f32_bf16<<<(int)((nH  + 255) / 256), 256, 0, stream>>>(hidden + nH, h1bf[0], (int)nH);

  const int layerBlocks  = RNN_HID / 64;                 // 32
  const int logitsBlocks = (RNN_VOCAB + 63) / 64;        // 157 (edge masked)
  const int embedBlocks  = (int)(nH / 256);              // 512

  for (int t = 0; t < RNN_SEQ; ++t) {
    const int p = t & 1, q = p ^ 1;  // ping-pong hidden buffers
    k_embed<<<embedBlocks, 256, 0, stream>>>(inputs + (size_t)t * RNN_BATCH, emb, xbf);
    // layer 0: A = [x_t | h0], K = 4096
    k_gemm_bf16<true, false><<<layerBlocks, 256, 0, stream>>>(
        xbf, h0bf[p], RNN_EMB, RNN_EMB + RNN_HID, RNN_HID,
        w0bf, b0, hf32, h0bf[q], RNN_HID);
    // layer 1: A = [h0_new | h1], K = 4096
    k_gemm_bf16<true, false><<<layerBlocks, 256, 0, stream>>>(
        h0bf[q], h1bf[p], RNN_HID, 2 * RNN_HID, RNN_HID,
        w1bf, b1, hf32 + nH, h1bf[q], RNN_HID);
    // logits: A = h1_new, K = 2048, N = 10000 (edge-guarded stores)
    k_gemm_bf16<false, true><<<logitsBlocks, 256, 0, stream>>>(
        h1bf[q], h1bf[q], RNN_HID, RNN_HID, RNN_VOCAB,
        wobf, bout, out + (size_t)t * nLog, (bf16_t*)nullptr, RNN_VOCAB);
  }

  // final_hidden (f32) appended after all logits.
  (void)hipMemcpyAsync(out + (size_t)RNN_SEQ * nLog, hf32, 2 * nH * sizeof(float),
                       hipMemcpyDeviceToDevice, stream);
}
